// GNNLF_91010357002570
// MI455X (gfx1250) — matrix operand order repaired and checked by reading, not compiled, not run on hardware
//
#include <hip/hip_runtime.h>
#include <hip/hip_bf16.h>

#define BBATCH 16
#define NATOM  96
#define HID    128
#define EFD    32
#define DIRD   256
#define NLAY   4

typedef __attribute__((ext_vector_type(16))) __bf16 v16bf;
typedef __attribute__((ext_vector_type(8)))  float  v8f;

union AFrag { v16bf v; __bf16 e[16]; };
union CFrag { v8f  v; float  e[8];  };

__device__ __forceinline__ __bf16 f2bf(float x) {
    union { float f; unsigned u; } a; a.f = x;
    unsigned r = a.u + 0x7FFFu + ((a.u >> 16) & 1u);   // round-to-nearest-even
    union { unsigned short s; __bf16 b; } c; c.s = (unsigned short)(r >> 16);
    return c.b;
}
__device__ __forceinline__ float bf2f(__bf16 b) {
    union { unsigned short s; __bf16 b; } c; c.b = b;
    union { unsigned u; float f; } a; a.u = ((unsigned)c.s) << 16;
    return a.f;
}
// fast silu: x * v_rcp_f32(1 + e^-x)  (avoids IEEE div refinement chain)
__device__ __forceinline__ float siluf(float x) {
    return x * __builtin_amdgcn_rcpf(1.0f + __expf(-x));
}
__device__ __forceinline__ float frsq(float x) { return __builtin_amdgcn_rsqf(x); }

// ---------------------------------------------------------------------------
// Pack a row-major (Kdim x Ndim) f32 weight into WMMA bf16 B-fragments.
// Fragment (kk,nn): lane L holds column n = nn*16 + (L%16);
// element e -> k = kk*32 + (L<16 ? 0 : 8) + (e<8 ? e : e+8)   (ISA 7.12.2, 16-bit B layout)
// ---------------------------------------------------------------------------
__global__ void k_pack(const float* __restrict__ src, __bf16* __restrict__ dst,
                       int Kdim, int Ndim) {
    int numN = Ndim >> 4;
    int total = (Kdim >> 5) * numN * 512;
    for (int idx = blockIdx.x * blockDim.x + threadIdx.x; idx < total;
         idx += gridDim.x * blockDim.x) {
        int f = idx >> 9, w = idx & 511;
        int lane = w >> 4, e = w & 15;
        int kk = f / numN, nn = f % numN;
        int k = kk * 32 + ((lane < 16) ? 0 : 8) + (e < 8 ? e : e + 8);
        int n = nn * 16 + (lane & 15);
        dst[idx] = f2bf(src[(size_t)k * Ndim + n]);
    }
}

// ---------------------------------------------------------------------------
// Geometry: d, ea (cutoff*pmask), ev (unit dirs * pmask), pad.
// dpm = d where pair valid else 1e4 (makes RBF underflow to exactly 0).
// ---------------------------------------------------------------------------
__global__ void k_geom(const int* __restrict__ z, const float* __restrict__ pos,
                       float* __restrict__ ea, float* __restrict__ ev,
                       float* __restrict__ dpm, float* __restrict__ pad) {
    int idx = blockIdx.x * blockDim.x + threadIdx.x;
    int total = BBATCH * NATOM * NATOM;
    if (idx >= total) return;
    int j = idx % NATOM, i = (idx / NATOM) % NATOM, b = idx / (NATOM * NATOM);
    if (idx < BBATCH * NATOM) pad[idx] = (z[idx] != 0) ? 1.0f : 0.0f;
    float pi_ = (z[b * NATOM + i] != 0) ? 1.0f : 0.0f;
    float pj_ = (z[b * NATOM + j] != 0) ? 1.0f : 0.0f;
    float pm = pi_ * pj_ * ((i != j) ? 1.0f : 0.0f);
    const float* Pi = pos + ((size_t)b * NATOM + i) * 3;
    const float* Pj = pos + ((size_t)b * NATOM + j) * 3;
    float rx = Pj[0] - Pi[0], ry = Pj[1] - Pi[1], rz = Pj[2] - Pi[2];
    float d = sqrtf(rx * rx + ry * ry + rz * rz + 1e-12f);
    float cut = (d < 10.0f) ? 1.0f : 0.0f;
    float eav = 0.5f * (cosf(3.14159265358979f * d * 0.1f) + 1.0f) * cut * pm;
    float inv = pm * __builtin_amdgcn_rcpf(d + 1e-12f);
    ea[idx] = eav;
    ev[(size_t)idx * 3 + 0] = rx * inv;
    ev[(size_t)idx * 3 + 1] = ry * inv;
    ev[(size_t)idx * 3 + 2] = rz * inv;
    dpm[idx] = (pm > 0.0f) ? d : 1.0e4f;
}

// ---------------------------------------------------------------------------
// s0 = LayerNorm(emb[z])  (no affine). One block per atom, 128 threads.
// ---------------------------------------------------------------------------
__global__ void k_s0(const int* __restrict__ z, const float* __restrict__ emb,
                     float* __restrict__ s0) {
    int bi = blockIdx.x, t = threadIdx.x;
    float x = emb[(size_t)z[bi] * HID + t];
    __shared__ float red[128];
    red[t] = x; __syncthreads();
    for (int s = 64; s > 0; s >>= 1) { if (t < s) red[t] += red[t + s]; __syncthreads(); }
    float mean = red[0] * (1.0f / HID); __syncthreads();
    float dx = x - mean;
    red[t] = dx * dx; __syncthreads();
    for (int s = 64; s > 0; s >>= 1) { if (t < s) red[t] += red[t + s]; __syncthreads(); }
    float var = red[0] * (1.0f / HID);
    s0[(size_t)bi * HID + t] = dx * frsq(var + 1e-5f);
}

// ---------------------------------------------------------------------------
// mask = silu(rbf(d) @ W_ef + b_ef) * ea   -> bf16 (B,N,N,HID)
// One block per (b,i); 6 tiles of 16 pairs; K=32 = exactly one wmma step.
// ---------------------------------------------------------------------------
__global__ void k_mask(const float* __restrict__ dpm, const float* __restrict__ ea,
                       const float* __restrict__ bef, const __bf16* __restrict__ wefp,
                       __bf16* __restrict__ mask) {
    int bi = blockIdx.x, t = threadIdx.x;
    int w = t >> 5, lane = t & 31;
    __shared__ __bf16 efs[16][EFD];
    __shared__ float eas[16];
    for (int jt = 0; jt < 6; jt++) {
        for (int q = t; q < 16 * EFD; q += 128) {
            int r = q >> 5, c = q & 31;
            float d = dpm[(size_t)bi * NATOM + jt * 16 + r];
            float ctr = (10.0f / 31.0f) * (float)c;
            float dd = d - ctr;
            efs[r][c] = f2bf(__expf(-10.0f * dd * dd));
        }
        if (t < 16) eas[t] = ea[(size_t)bi * NATOM + jt * 16 + t];
        __syncthreads();
        int row = lane & 15, kb = (lane < 16) ? 0 : 8;
        AFrag a;
        #pragma unroll
        for (int e = 0; e < 16; e++) { int k = kb + (e < 8 ? e : e + 8); a.e[e] = efs[row][k]; }
        #pragma unroll
        for (int nt = 0; nt < 2; nt++) {
            int n0 = w * 32 + nt * 16;
            AFrag bfr; CFrag acc;
            #pragma unroll
            for (int q = 0; q < 8; q++) acc.e[q] = 0.0f;
            const __bf16* src = wefp + ((size_t)(n0 >> 4) * 32 + lane) * 16;
            #pragma unroll
            for (int e = 0; e < 16; e++) bfr.e[e] = src[e];
            acc.v = __builtin_amdgcn_wmma_f32_16x16x32_bf16(false, a.v, false, bfr.v,
                                                            (short)0, acc.v, false, false);
            int col = n0 + (lane & 15);
            float bv = bef[col];
            #pragma unroll
            for (int r8 = 0; r8 < 8; r8++) {
                int M = r8 + ((lane < 16) ? 0 : 8);
                float val = siluf(acc.e[r8] + bv) * eas[M];
                mask[((size_t)bi * NATOM + jt * 16 + M) * HID + col] = f2bf(val);
            }
        }
        __syncthreads();
    }
}

// s[b,i,h] = s0[b,i,h] + sum_j mask[b,i,j,h] * s0[b,j,h]   (elementwise in h)
__global__ void k_neigh(const __bf16* __restrict__ mask, const float* __restrict__ s0,
                        float* __restrict__ s) {
    int bi = blockIdx.x, t = threadIdx.x, b = bi / NATOM;
    float acc = 0.0f;
    const __bf16* mrow = mask + (size_t)bi * NATOM * HID + t;
    const float* s0b = s0 + (size_t)b * NATOM * HID + t;
    for (int j = 0; j < NATOM; j++)
        acc += bf2f(mrow[(size_t)j * HID]) * s0b[(size_t)j * HID];
    s[(size_t)bi * HID + t] = s0[(size_t)bi * HID + t] + acc;
}

// ---------------------------------------------------------------------------
// Generic (rows x 128) @ (128 x 128) WMMA GEMM, packed weights.
// mode 0: out = A@W + b ; mode 2: out = silu(LN(A@W + b))
// One block per 16 rows; 4 waves x 2 n-tiles x 4 k-steps.
// ---------------------------------------------------------------------------
__global__ void k_gemm128(const float* __restrict__ A, const __bf16* __restrict__ Wp,
                          const float* __restrict__ bias, float* __restrict__ out,
                          int mode) {
    int tile = blockIdx.x, t = threadIdx.x;
    int w = t >> 5, lane = t & 31;
    __shared__ __bf16 as[16][HID];
    __shared__ float  os[16][HID];
    for (int q = t; q < 16 * HID; q += 128) {
        int r = q >> 7, c = q & 127;
        as[r][c] = f2bf(A[((size_t)tile * 16 + r) * HID + c]);
    }
    __syncthreads();
    int row = lane & 15, kb = (lane < 16) ? 0 : 8;
    CFrag acc[2];
    #pragma unroll
    for (int nt = 0; nt < 2; nt++)
        #pragma unroll
        for (int q = 0; q < 8; q++) acc[nt].e[q] = 0.0f;
    for (int kk = 0; kk < 4; kk++) {
        AFrag a;
        #pragma unroll
        for (int e = 0; e < 16; e++) { int k = kk * 32 + kb + (e < 8 ? e : e + 8); a.e[e] = as[row][k]; }
        #pragma unroll
        for (int nt = 0; nt < 2; nt++) {
            int fragIdx = kk * 8 + ((w * 32 + nt * 16) >> 4);
            AFrag bfr;
            const __bf16* src = Wp + ((size_t)fragIdx * 32 + lane) * 16;
            #pragma unroll
            for (int e = 0; e < 16; e++) bfr.e[e] = src[e];
            acc[nt].v = __builtin_amdgcn_wmma_f32_16x16x32_bf16(false, a.v, false, bfr.v,
                                                                (short)0, acc[nt].v, false, false);
        }
    }
    #pragma unroll
    for (int nt = 0; nt < 2; nt++) {
        int col = w * 32 + nt * 16 + (lane & 15);
        float bv = bias ? bias[col] : 0.0f;
        #pragma unroll
        for (int r8 = 0; r8 < 8; r8++) {
            int M = r8 + ((lane < 16) ? 0 : 8);
            float val = acc[nt].e[r8] + bv;
            if (mode == 2) os[M][col] = val;
            else           out[((size_t)tile * 16 + M) * HID + col] = val;
        }
    }
    if (mode == 2) {
        __syncthreads();
        if (t < 16) {
            float m = 0.0f;
            for (int c = 0; c < HID; c++) m += os[t][c];
            m *= (1.0f / HID);
            float v = 0.0f;
            for (int c = 0; c < HID; c++) { float dx = os[t][c] - m; v += dx * dx; }
            float inv = frsq(v * (1.0f / HID) + 1e-5f);
            for (int c = 0; c < HID; c++)
                out[((size_t)tile * 16 + t) * HID + c] = siluf((os[t][c] - m) * inv);
        }
    }
}

// v[b,i,k,h] = sum_j mask * s1[b,j,h] * ev[b,i,j,k]
__global__ void k_v(const __bf16* __restrict__ mask, const float* __restrict__ s1,
                    const float* __restrict__ ev, float* __restrict__ v) {
    int bi = blockIdx.x, t = threadIdx.x, b = bi / NATOM;
    float a0 = 0, a1 = 0, a2 = 0;
    for (int j = 0; j < NATOM; j++) {
        float m = bf2f(mask[((size_t)bi * NATOM + j) * HID + t]) *
                  s1[((size_t)b * NATOM + j) * HID + t];
        const float* e3 = ev + ((size_t)bi * NATOM + j) * 3;
        a0 += m * e3[0]; a1 += m * e3[1]; a2 += m * e3[2];
    }
    v[((size_t)bi * 3 + 0) * HID + t] = a0;
    v[((size_t)bi * 3 + 1) * HID + t] = a1;
    v[((size_t)bi * 3 + 2) * HID + t] = a2;
}

// ---------------------------------------------------------------------------
// Fused directional MLP: dirf = [dir2|dir3]*ea (LDS, bf16) ->
// h1 = silu(dirf@W_d1+b1) (LDS) -> dp = silu(h1@W_d2+b2) -> mask *= dp.
// One block per (b,i); 6 tiles of 16 pairs; 48 wmma / wave / tile.
// ---------------------------------------------------------------------------
__global__ void k_dp(const float* __restrict__ v, const float* __restrict__ qv,
                     const float* __restrict__ kv, const float* __restrict__ ea,
                     const float* __restrict__ ev, const float* __restrict__ bd1,
                     const float* __restrict__ bd2, const __bf16* __restrict__ wd1p,
                     const __bf16* __restrict__ wd2p, __bf16* __restrict__ mask) {
    int bi = blockIdx.x, t = threadIdx.x;
    int w = t >> 5, lane = t & 31, b = bi / NATOM;
    __shared__ float vv[3][HID], kvv[3][HID];
    __shared__ float evs[16][3], eas[16];
    __shared__ __bf16 dirf[16][DIRD];
    __shared__ __bf16 h1[16][DIRD];
    for (int q = t; q < 3 * HID; q += 128) {
        vv[q / HID][q % HID]  = v[(size_t)bi * 3 * HID + q];
        kvv[q / HID][q % HID] = kv[(size_t)bi * 3 * HID + q];
    }
    __syncthreads();
    for (int jt = 0; jt < 6; jt++) {
        if (t < 16) {
            int j = jt * 16 + t;
            eas[t] = ea[(size_t)bi * NATOM + j];
            for (int k = 0; k < 3; k++) evs[t][k] = ev[((size_t)bi * NATOM + j) * 3 + k];
        }
        __syncthreads();
        int h = t;
        for (int r = 0; r < 16; r++) {
            int j = jt * 16 + r;
            float d2 = vv[0][h] * evs[r][0] + vv[1][h] * evs[r][1] + vv[2][h] * evs[r][2];
            const float* qrow = qv + ((size_t)(b * NATOM + j)) * 3 * HID + h;
            float d3 = qrow[0] * kvv[0][h] + qrow[HID] * kvv[1][h] + qrow[2 * HID] * kvv[2][h];
            dirf[r][h]       = f2bf(d2 * eas[r]);
            dirf[r][HID + h] = f2bf(d3 * eas[r]);
        }
        __syncthreads();
        int row = lane & 15, kb = (lane < 16) ? 0 : 8;
        { // GEMM1: 16x256 @ 256x256, wave w -> cols [w*64, w*64+64)
            CFrag acc[4];
            #pragma unroll
            for (int nt = 0; nt < 4; nt++)
                #pragma unroll
                for (int q = 0; q < 8; q++) acc[nt].e[q] = 0.0f;
            for (int kk = 0; kk < 8; kk++) {
                AFrag a;
                #pragma unroll
                for (int e = 0; e < 16; e++) { int k = kk * 32 + kb + (e < 8 ? e : e + 8); a.e[e] = dirf[row][k]; }
                #pragma unroll
                for (int nt = 0; nt < 4; nt++) {
                    int fragIdx = kk * 16 + (w * 4 + nt);
                    AFrag bfr;
                    const __bf16* src = wd1p + ((size_t)fragIdx * 32 + lane) * 16;
                    #pragma unroll
                    for (int e = 0; e < 16; e++) bfr.e[e] = src[e];
                    acc[nt].v = __builtin_amdgcn_wmma_f32_16x16x32_bf16(false, a.v, false, bfr.v,
                                                                        (short)0, acc[nt].v, false, false);
                }
            }
            #pragma unroll
            for (int nt = 0; nt < 4; nt++) {
                int col = w * 64 + nt * 16 + (lane & 15);
                float bv = bd1[col];
                #pragma unroll
                for (int r8 = 0; r8 < 8; r8++) {
                    int M = r8 + ((lane < 16) ? 0 : 8);
                    h1[M][col] = f2bf(siluf(acc[nt].e[r8] + bv));
                }
            }
        }
        __syncthreads();
        { // GEMM2: 16x256 @ 256x128, wave w -> cols [w*32, w*32+32)
            CFrag acc[2];
            #pragma unroll
            for (int nt = 0; nt < 2; nt++)
                #pragma unroll
                for (int q = 0; q < 8; q++) acc[nt].e[q] = 0.0f;
            for (int kk = 0; kk < 8; kk++) {
                AFrag a;
                #pragma unroll
                for (int e = 0; e < 16; e++) { int k = kk * 32 + kb + (e < 8 ? e : e + 8); a.e[e] = h1[row][k]; }
                #pragma unroll
                for (int nt = 0; nt < 2; nt++) {
                    int fragIdx = kk * 8 + (w * 2 + nt);
                    AFrag bfr;
                    const __bf16* src = wd2p + ((size_t)fragIdx * 32 + lane) * 16;
                    #pragma unroll
                    for (int e = 0; e < 16; e++) bfr.e[e] = src[e];
                    acc[nt].v = __builtin_amdgcn_wmma_f32_16x16x32_bf16(false, a.v, false, bfr.v,
                                                                        (short)0, acc[nt].v, false, false);
                }
            }
            #pragma unroll
            for (int nt = 0; nt < 2; nt++) {
                int col = w * 32 + nt * 16 + (lane & 15);
                float bv = bd2[col];
                #pragma unroll
                for (int r8 = 0; r8 < 8; r8++) {
                    int M = r8 + ((lane < 16) ? 0 : 8);
                    float dp = siluf(acc[nt].e[r8] + bv);
                    size_t mi = ((size_t)bi * NATOM + jt * 16 + M) * HID + col;
                    mask[mi] = f2bf(bf2f(mask[mi]) * dp);
                }
            }
        }
        __syncthreads();
    }
}

// s[b,i,h] += sum_j mask[b,i,j,h] * sl[b,j,h]
__global__ void k_agg(const __bf16* __restrict__ mask, const float* __restrict__ sl,
                      float* __restrict__ s) {
    int bi = blockIdx.x, t = threadIdx.x, b = bi / NATOM;
    float acc = 0.0f;
    for (int j = 0; j < NATOM; j++)
        acc += bf2f(mask[((size_t)bi * NATOM + j) * HID + t]) *
               sl[((size_t)b * NATOM + j) * HID + t];
    s[(size_t)bi * HID + t] += acc;
}

// out[b] = (sum_i s[b,i,:]*pad) . W_out + b_out
__global__ void k_read(const float* __restrict__ s, const float* __restrict__ pad,
                       const float* __restrict__ wout, const float* __restrict__ bout,
                       float* __restrict__ out) {
    int b = blockIdx.x, t = threadIdx.x;
    float acc = 0.0f;
    for (int i = 0; i < NATOM; i++)
        acc += s[((size_t)(b * NATOM + i)) * HID + t] * pad[b * NATOM + i];
    float p = acc * wout[t];
    __shared__ float red[128];
    red[t] = p; __syncthreads();
    for (int st = 64; st > 0; st >>= 1) { if (t < st) red[t] += red[t + st]; __syncthreads(); }
    if (t == 0) out[b] = red[0] + bout[0];
}

// ---------------------------------------------------------------------------
extern "C" void kernel_launch(void* const* d_in, const int* in_sizes, int n_in,
                              void* d_out, int out_size, void* d_ws, size_t ws_size,
                              hipStream_t stream) {
    (void)in_sizes; (void)n_in; (void)out_size; (void)ws_size;
    const int*   z     = (const int*)  d_in[0];
    const float* pos   = (const float*)d_in[1];
    const float* emb   = (const float*)d_in[2];
    const float* W_ef  = (const float*)d_in[3];
    const float* b_ef  = (const float*)d_in[4];
    const float* W_s2v = (const float*)d_in[5];
    const float* b_s2v = (const float*)d_in[6];
    const float* W_q   = (const float*)d_in[7];
    const float* W_k   = (const float*)d_in[8];
    const float* W_d1  = (const float*)d_in[9];
    const float* b_d1  = (const float*)d_in[10];
    const float* W_d2  = (const float*)d_in[11];
    const float* b_d2  = (const float*)d_in[12];
    const float* W_int = (const float*)d_in[13];
    const float* b_int = (const float*)d_in[14];
    const float* W_out = (const float*)d_in[15];
    const float* b_out = (const float*)d_in[16];

    char* ws = (char*)d_ws;
    size_t off = 0;
    auto alloc = [&](size_t bytes) -> size_t {
        size_t r = off; off += (bytes + 255) & ~(size_t)255; return r;
    };
    const size_t BN = (size_t)BBATCH * NATOM;
    const size_t BNN = BN * NATOM;
    float* pad_b = (float*)(ws + alloc(BN * 4));
    float* ea_b  = (float*)(ws + alloc(BNN * 4));
    float* ev_b  = (float*)(ws + alloc(BNN * 3 * 4));
    float* dpm_b = (float*)(ws + alloc(BNN * 4));
    float* s0_b  = (float*)(ws + alloc(BN * HID * 4));
    float* s_b   = (float*)(ws + alloc(BN * HID * 4));
    float* s1_b  = (float*)(ws + alloc(BN * HID * 4));
    float* v_b   = (float*)(ws + alloc(BN * 3 * HID * 4));
    float* qv_b  = (float*)(ws + alloc(BN * 3 * HID * 4));
    float* kv_b  = (float*)(ws + alloc(BN * 3 * HID * 4));
    __bf16* mask_b = (__bf16*)(ws + alloc(BNN * HID * 2));
    __bf16* wefp  = (__bf16*)(ws + alloc((size_t)(EFD/32)*(HID/16)*512 * 2));
    __bf16* ws2vp = (__bf16*)(ws + alloc((size_t)4*8*512 * 2));
    __bf16* wqp   = (__bf16*)(ws + alloc((size_t)4*8*512 * 2));
    __bf16* wkp   = (__bf16*)(ws + alloc((size_t)4*8*512 * 2));
    __bf16* wd1p  = (__bf16*)(ws + alloc((size_t)8*16*512 * 2));
    __bf16* wd2p  = (__bf16*)(ws + alloc((size_t)8*8*512 * 2));
    __bf16* wintp = (__bf16*)(ws + alloc((size_t)NLAY*4*8*512 * 2));

    auto pack = [&](const float* src, __bf16* dst, int K, int Nd) {
        int total = (K / 32) * (Nd / 16) * 512;
        k_pack<<<(total + 255) / 256, 256, 0, stream>>>(src, dst, K, Nd);
    };
    pack(W_ef,  wefp,  EFD,  HID);
    pack(W_s2v, ws2vp, HID,  HID);
    pack(W_q,   wqp,   HID,  HID);
    pack(W_k,   wkp,   HID,  HID);
    pack(W_d1,  wd1p,  DIRD, DIRD);
    pack(W_d2,  wd2p,  DIRD, HID);
    for (int l = 0; l < NLAY; l++)
        pack(W_int + (size_t)l * HID * HID, wintp + (size_t)l * 4 * 8 * 512, HID, HID);

    int totPairs = BBATCH * NATOM * NATOM;
    k_geom<<<(totPairs + 255) / 256, 256, 0, stream>>>(z, pos, ea_b, ev_b, dpm_b, pad_b);
    k_s0<<<(int)BN, 128, 0, stream>>>(z, emb, s0_b);
    k_mask<<<(int)BN, 128, 0, stream>>>(dpm_b, ea_b, b_ef, wefp, mask_b);
    k_neigh<<<(int)BN, 128, 0, stream>>>(mask_b, s0_b, s_b);
    k_gemm128<<<(int)(BN / 16), 128, 0, stream>>>(s_b, ws2vp, b_s2v, s1_b, 2);
    k_v<<<(int)BN, 128, 0, stream>>>(mask_b, s1_b, ev_b, v_b);
    k_gemm128<<<(int)(BN * 3 / 16), 128, 0, stream>>>(v_b, wqp, nullptr, qv_b, 0);
    k_gemm128<<<(int)(BN * 3 / 16), 128, 0, stream>>>(v_b, wkp, nullptr, kv_b, 0);
    k_dp<<<(int)BN, 128, 0, stream>>>(v_b, qv_b, kv_b, ea_b, ev_b, b_d1, b_d2,
                                      wd1p, wd2p, mask_b);
    for (int l = 0; l < NLAY; l++) {
        k_gemm128<<<(int)(BN / 16), 128, 0, stream>>>(
            s_b, wintp + (size_t)l * 4 * 8 * 512, b_int + (size_t)l * HID, s1_b, 2);
        k_agg<<<(int)BN, 128, 0, stream>>>(mask_b, s1_b, s_b);
    }
    k_read<<<BBATCH, 128, 0, stream>>>(s_b, pad_b, W_out, b_out, (float*)d_out);
}
